// Classifier_GNN_concat_23682449670461
// MI455X (gfx1250) — matrix operand order repaired and checked by reading, not compiled, not run on hardware
//
#include <hip/hip_runtime.h>
#include <hip/hip_bf16.h>

typedef __attribute__((ext_vector_type(16))) __bf16 bv16;
typedef __attribute__((ext_vector_type(8)))  __bf16 bv8;
typedef __attribute__((ext_vector_type(8)))  float  fv8;

// ---------------------------------------------------------------- utilities
__global__ void k_fill_f32(float* __restrict__ p, float v, long long n) {
  long long i = (long long)blockIdx.x * blockDim.x + threadIdx.x;
  if (i < n) p[i] = v;
}

// per-channel batch mean / rsqrt(var+eps) over the node dimension
__global__ void k_bn_stats(const float* __restrict__ X, int N, int F,
                           float* __restrict__ mean, float* __restrict__ rstd) {
  int c = blockIdx.x * blockDim.x + threadIdx.x;
  if (c >= F) return;
  float s = 0.f, ss = 0.f;
  for (int n = 0; n < N; ++n) {
    float v = X[(long long)n * F + c];
    s += v; ss += v * v;
  }
  float m = s / (float)N;
  float var = ss / (float)N - m * m;
  mean[c] = m;
  rstd[c] = rsqrtf(var + 1e-5f);
}

// BN apply (+optional relu) fused with f32->bf16 convert and K zero-padding
__global__ void k_bn_apply_bf16(const float* __restrict__ X,
                                const float* __restrict__ mean,
                                const float* __restrict__ rstd,
                                const float* __restrict__ g,
                                const float* __restrict__ b,
                                __bf16* __restrict__ Y,
                                int N, int F, int Kpad, int relu) {
  long long i = (long long)blockIdx.x * blockDim.x + threadIdx.x;
  long long tot = (long long)N * Kpad;
  if (i >= tot) return;
  int n = (int)(i / Kpad), k = (int)(i % Kpad);
  float v = 0.f;
  if (k < F) {
    v = (X[(long long)n * F + k] - mean[k]) * rstd[k] * g[k] + b[k];
    if (relu) v = fmaxf(v, 0.f);
  }
  Y[i] = (__bf16)v;
}

// pack [K x Nh] Wl and Wr (f32, row-major) into fused transposed bf16
// Bt[(2*Nh) x Kpad]: row n<Nh -> Wl column n, row n>=Nh -> Wr column n-Nh
__global__ void k_pack_w(const float* __restrict__ Wl, const float* __restrict__ Wr,
                         __bf16* __restrict__ Bt, int K, int Kpad, int Nh) {
  long long i = (long long)blockIdx.x * blockDim.x + threadIdx.x;
  long long tot = (long long)(2 * Nh) * Kpad;
  if (i >= tot) return;
  int nn = (int)(i / Kpad), k = (int)(i % Kpad);
  float v = 0.f;
  if (k < K) v = (nn < Nh) ? Wl[(long long)k * Nh + nn]
                           : Wr[(long long)k * Nh + (nn - Nh)];
  Bt[i] = (__bf16)v;
}

// ---------------------------------------------------------------- WMMA GEMM
// C[M x Nout] (f32, row-major) = A[M x K] (bf16 row-major) * Bt^T
// Bt is [Nout x K] bf16 row-major (B column-major). K % 32 == 0,
// M % 64 == 0, Nout % 32 == 0.
// Register-blocked: one wave32 computes a 64x32 C macro-tile
// (4 M-subtiles x 2 N-subtiles = 8 WMMA accumulators) so each k-step
// loads 4 A + 2 B fragments (6 KB) for 8 WMMAs (~22 FLOP/byte from L2).
__global__ void __launch_bounds__(256)
k_gemm_bf16(const __bf16* __restrict__ A, const __bf16* __restrict__ Bt,
            float* __restrict__ C, int M, int K, int Nout) {
  int wid  = (int)(((long long)blockIdx.x * blockDim.x + threadIdx.x) >> 5);
  int lane = threadIdx.x & 31;
  int tn = Nout >> 5;            // 32-col macro tiles
  int tm = M >> 6;               // 64-row macro tiles
  if (wid >= tm * tn) return;
  int mt = wid / tn, nt = wid % tn;
  int lo = lane & 15, hi = lane >> 4;

  // A fragment (16-bit A 16x32): lane row = base+lo; lanes 0-15 carry
  // K = k0..k0+15 contiguous, lanes 16-31 carry K = k0+16..k0+31.
  const __bf16* Ar0 = A + (long long)(mt * 64 + 0  + lo) * K + 16 * hi;
  const __bf16* Ar1 = A + (long long)(mt * 64 + 16 + lo) * K + 16 * hi;
  const __bf16* Ar2 = A + (long long)(mt * 64 + 32 + lo) * K + 16 * hi;
  const __bf16* Ar3 = A + (long long)(mt * 64 + 48 + lo) * K + 16 * hi;
  // B fragment (16-bit B 32x16): lane col = base+lo; lanes 0-15 carry
  // K = k0..k0+15 contiguous, lanes 16-31 carry K = k0+16..k0+31.
  const __bf16* Br0 = Bt + (long long)(nt * 32 + 0  + lo) * K + 16 * hi;
  const __bf16* Br1 = Bt + (long long)(nt * 32 + 16 + lo) * K + 16 * hi;

  fv8 acc00 = {0.f,0.f,0.f,0.f,0.f,0.f,0.f,0.f};
  fv8 acc01 = acc00, acc10 = acc00, acc11 = acc00;
  fv8 acc20 = acc00, acc21 = acc00, acc30 = acc00, acc31 = acc00;

  for (int k0 = 0; k0 < K; k0 += 32) {
    bv8 a0l = *(const bv8*)(Ar0 + k0);
    bv8 a0h = *(const bv8*)(Ar0 + k0 + 8);
    bv8 a1l = *(const bv8*)(Ar1 + k0);
    bv8 a1h = *(const bv8*)(Ar1 + k0 + 8);
    bv8 a2l = *(const bv8*)(Ar2 + k0);
    bv8 a2h = *(const bv8*)(Ar2 + k0 + 8);
    bv8 a3l = *(const bv8*)(Ar3 + k0);
    bv8 a3h = *(const bv8*)(Ar3 + k0 + 8);
    bv16 b0 = *(const bv16*)(Br0 + k0);
    bv16 b1 = *(const bv16*)(Br1 + k0);
    bv16 a0 = __builtin_shufflevector(a0l, a0h, 0,1,2,3,4,5,6,7,8,9,10,11,12,13,14,15);
    bv16 a1 = __builtin_shufflevector(a1l, a1h, 0,1,2,3,4,5,6,7,8,9,10,11,12,13,14,15);
    bv16 a2 = __builtin_shufflevector(a2l, a2h, 0,1,2,3,4,5,6,7,8,9,10,11,12,13,14,15);
    bv16 a3 = __builtin_shufflevector(a3l, a3h, 0,1,2,3,4,5,6,7,8,9,10,11,12,13,14,15);
    acc00 = __builtin_amdgcn_wmma_f32_16x16x32_bf16(false, a0, false, b0, (short)0, acc00, false, false);
    acc01 = __builtin_amdgcn_wmma_f32_16x16x32_bf16(false, a0, false, b1, (short)0, acc01, false, false);
    acc10 = __builtin_amdgcn_wmma_f32_16x16x32_bf16(false, a1, false, b0, (short)0, acc10, false, false);
    acc11 = __builtin_amdgcn_wmma_f32_16x16x32_bf16(false, a1, false, b1, (short)0, acc11, false, false);
    acc20 = __builtin_amdgcn_wmma_f32_16x16x32_bf16(false, a2, false, b0, (short)0, acc20, false, false);
    acc21 = __builtin_amdgcn_wmma_f32_16x16x32_bf16(false, a2, false, b1, (short)0, acc21, false, false);
    acc30 = __builtin_amdgcn_wmma_f32_16x16x32_bf16(false, a3, false, b0, (short)0, acc30, false, false);
    acc31 = __builtin_amdgcn_wmma_f32_16x16x32_bf16(false, a3, false, b1, (short)0, acc31, false, false);
  }

  // C/D layout: VGPR r -> row (hi*8 + r) within subtile, col = lo
  fv8 accs[4][2] = {{acc00, acc01}, {acc10, acc11},
                    {acc20, acc21}, {acc30, acc31}};
#pragma unroll
  for (int ms = 0; ms < 4; ++ms) {
#pragma unroll
    for (int ns = 0; ns < 2; ++ns) {
      int row0 = mt * 64 + ms * 16 + hi * 8;
      int col  = nt * 32 + ns * 16 + lo;
#pragma unroll
      for (int r = 0; r < 8; ++r)
        C[(long long)(row0 + r) * Nout + col] = accs[ms][ns][r];
    }
  }
}

// ---------------------------------------------------------------- GAT edges
__global__ void k_deg(const int* __restrict__ dst, int E0, int N,
                      float* __restrict__ deg) {
  int e = blockIdx.x * blockDim.x + threadIdx.x;
  if (e >= E0 + N) return;
  int d = (e < E0) ? dst[e] : (e - E0);
  atomicAdd(&deg[d], 1.0f);
}

// wave per (edge, head): e = att . leaky_relu(xl[src]+xr[dst]); segment max
__global__ void k_edge_score(const float* __restrict__ XLR,
                             const float* __restrict__ att,
                             const int* __restrict__ src,
                             const int* __restrict__ dst,
                             int E0, int N, int H, int C,
                             float* __restrict__ esc, float* __restrict__ emax) {
  int gw   = (int)(((long long)blockIdx.x * blockDim.x + threadIdx.x) >> 5);
  int lane = threadIdx.x & 31;
  int total = (E0 + N) * H;
  if (gw >= total) return;
  int e = gw / H, h = gw % H;
  int s = (e < E0) ? src[e] : (e - E0);
  int d = (e < E0) ? dst[e] : (e - E0);
  int HC = H * C;
  const float* xl = XLR + (long long)s * (2 * HC) + h * C;
  const float* xr = XLR + (long long)d * (2 * HC) + HC + h * C;
  const float* av = att + h * C;
  float sum = 0.f;
  for (int c = lane; c < C; c += 32) {
    float z = xl[c] + xr[c];
    z = (z > 0.f) ? z : 0.2f * z;          // leaky_relu slope 0.2
    sum += z * av[c];
  }
  for (int off = 16; off > 0; off >>= 1) sum += __shfl_xor(sum, off, 32);
  if (lane == 0) {
    esc[gw] = sum;
    atomicMax(&emax[(long long)d * H + h], sum);  // global_atomic_max_num_f32
  }
}

__global__ void k_edge_exp(const int* __restrict__ dst, int E0, int N, int H,
                           float* __restrict__ esc, const float* __restrict__ emax,
                           float* __restrict__ denom) {
  int i = blockIdx.x * blockDim.x + threadIdx.x;
  int total = (E0 + N) * H;
  if (i >= total) return;
  int e = i / H, h = i % H;
  int d = (e < E0) ? dst[e] : (e - E0);
  float ex = expf(esc[i] - emax[(long long)d * H + h]);
  esc[i] = ex;                               // overwrite score with exp
  atomicAdd(&denom[(long long)d * H + h], ex);
}

// thread per (edge, channel): agg[dst] += alpha * xl[src]
__global__ void k_edge_agg(const float* __restrict__ XLR,
                           const float* __restrict__ esc,
                           const float* __restrict__ denom,
                           const int* __restrict__ src,
                           const int* __restrict__ dst,
                           int E0, int N, int H, int C, float* __restrict__ agg) {
  long long i = (long long)blockIdx.x * blockDim.x + threadIdx.x;
  int HC = H * C;
  long long total = (long long)(E0 + N) * HC;
  if (i >= total) return;
  int e  = (int)(i / HC), hc = (int)(i % HC);
  int h  = hc / C;
  int s = (e < E0) ? src[e] : (e - E0);
  int d = (e < E0) ? dst[e] : (e - E0);
  float alpha = esc[(long long)e * H + h] / denom[(long long)d * H + h];
  atomicAdd(&agg[(long long)d * HC + hc],
            alpha * XLR[(long long)s * (2 * HC) + hc]);
}

__global__ void k_finalize(const float* __restrict__ agg,
                           const float* __restrict__ deg,
                           const float* __restrict__ bias,
                           float* __restrict__ out, int N, int HC, int relu) {
  long long i = (long long)blockIdx.x * blockDim.x + threadIdx.x;
  if (i >= (long long)N * HC) return;
  int n = (int)(i / HC), c = (int)(i % HC);
  float v = agg[i] / deg[n] + bias[c];
  if (relu) v = fmaxf(v, 0.f);
  out[i] = v;
}

// ---------------------------------------------------------------- head
__global__ void k_head(const float* __restrict__ Hf, const float* __restrict__ Wh,
                       const float* __restrict__ bh, float* __restrict__ pred,
                       int N, int C) {
  int gw   = (int)(((long long)blockIdx.x * blockDim.x + threadIdx.x) >> 5);
  int lane = threadIdx.x & 31;
  if (gw >= N) return;
  const float* row = Hf + (long long)gw * C;
  float s = 0.f;
  for (int c = lane; c < C; c += 32) s += row[c] * Wh[c];
  for (int off = 16; off > 0; off >>= 1) s += __shfl_xor(s, off, 32);
  if (lane == 0) pred[gw] = 1.0f / (1.0f + expf(-(s + bh[0])));
}

__global__ void k_gather(const float* __restrict__ pred, const float* __restrict__ y,
                         const int* __restrict__ tidx, float* __restrict__ out,
                         int NT) {
  int i = blockIdx.x * blockDim.x + threadIdx.x;
  if (i >= NT) return;
  int t = tidx[i];
  out[i]      = pred[t];
  out[NT + i] = y[t];
}

// ---------------------------------------------------------------- host
static inline int nblk(long long n, int bs) { return (int)((n + bs - 1) / bs); }

extern "C" void kernel_launch(void* const* d_in, const int* in_sizes, int n_in,
                              void* d_out, int out_size, void* d_ws, size_t ws_size,
                              hipStream_t stream) {
  (void)in_sizes; (void)n_in; (void)out_size; (void)ws_size;
  const float* x     = (const float*)d_in[0];
  const int*   ei    = (const int*)  d_in[1];
  const float* y     = (const float*)d_in[2];
  const int*   tidx  = (const int*)  d_in[3];
  const float* bn0_g = (const float*)d_in[4];
  const float* bn0_b = (const float*)d_in[5];
  const float* W1l = (const float*)d_in[6];  const float* W1r = (const float*)d_in[7];
  const float* a1  = (const float*)d_in[8];  const float* b1  = (const float*)d_in[9];
  const float* bn1_g = (const float*)d_in[10]; const float* bn1_b = (const float*)d_in[11];
  const float* W2l = (const float*)d_in[12]; const float* W2r = (const float*)d_in[13];
  const float* a2  = (const float*)d_in[14]; const float* b2  = (const float*)d_in[15];
  const float* bn2_g = (const float*)d_in[16]; const float* bn2_b = (const float*)d_in[17];
  const float* W3l = (const float*)d_in[18]; const float* W3r = (const float*)d_in[19];
  const float* a3  = (const float*)d_in[20]; const float* b3  = (const float*)d_in[21];
  const float* bn3_g = (const float*)d_in[22]; const float* bn3_b = (const float*)d_in[23];
  const float* W4l = (const float*)d_in[24]; const float* W4r = (const float*)d_in[25];
  const float* a4  = (const float*)d_in[26]; const float* b4  = (const float*)d_in[27];
  const float* Wh  = (const float*)d_in[28]; const float* bh  = (const float*)d_in[29];

  const int N = 8000, E0 = 64000, F = 3201, NT = 4000;
  const int KP1 = 3232;                       // 3201 padded to mult of 32
  const int* src = ei;
  const int* dst = ei + E0;

  // ---- workspace carve (256B aligned) ----
  char* wp = (char*)d_ws;
  auto carve = [&](size_t bytes) -> void* {
    void* p = (void*)wp;
    wp += (bytes + 255) & ~(size_t)255;
    return p;
  };
  __bf16* hbf   = (__bf16*)carve((size_t)N * KP1 * 2);        // bf16 A buffer
  __bf16* Bt    = (__bf16*)carve((size_t)2048 * KP1 * 2);     // packed weights
  float*  gemmC = (float*) carve((size_t)N * 2048 * 4);       // XL|XR
  float*  hcur  = (float*) carve((size_t)N * 1024 * 4);       // layer activ.
  float*  agg   = (float*) carve((size_t)N * 1024 * 4);
  float*  meanb = (float*) carve(4096 * 4);
  float*  rstdb = (float*) carve(4096 * 4);
  float*  esc   = (float*) carve((size_t)(E0 + N) * 2 * 4);
  float*  emax  = (float*) carve((size_t)N * 2 * 4);
  float*  denom = (float*) carve((size_t)N * 2 * 4);
  float*  deg   = (float*) carve((size_t)N * 4);
  float*  pred  = (float*) carve((size_t)N * 4);

  // in-degree (with self loops) — identical for every layer
  hipMemsetAsync(deg, 0, (size_t)N * 4, stream);
  k_deg<<<nblk(E0 + N, 256), 256, 0, stream>>>(dst, E0, N, deg);

  // BN(+relu) -> bf16 pad -> pack weights -> fused WMMA GEMM producing XL|XR
  auto dense = [&](const float* Hin, int Fin, int Kpad, const float* g,
                   const float* bb, int relu_in, const float* Wl,
                   const float* Wr, int Nh) {
    k_bn_stats<<<nblk(Fin, 256), 256, 0, stream>>>(Hin, N, Fin, meanb, rstdb);
    k_bn_apply_bf16<<<nblk((long long)N * Kpad, 256), 256, 0, stream>>>(
        Hin, meanb, rstdb, g, bb, hbf, N, Fin, Kpad, relu_in);
    k_pack_w<<<nblk((long long)2 * Nh * Kpad, 256), 256, 0, stream>>>(
        Wl, Wr, Bt, Fin, Kpad, Nh);
    long long waves = (long long)(N / 64) * ((2 * Nh) / 32);
    k_gemm_bf16<<<nblk(waves * 32, 256), 256, 0, stream>>>(
        hbf, Bt, gemmC, N, Kpad, 2 * Nh);
  };

  // GATv2 attention + mean aggregation from XL|XR in gemmC
  auto gat = [&](const float* att, const float* bias, int H, int C,
                 float* hout, int relu_out) {
    int HC = H * C;
    k_fill_f32<<<nblk((long long)N * H, 256), 256, 0, stream>>>(
        emax, -3.0e38f, (long long)N * H);
    hipMemsetAsync(denom, 0, (size_t)N * H * 4, stream);
    k_edge_score<<<nblk((long long)(E0 + N) * H * 32, 256), 256, 0, stream>>>(
        gemmC, att, src, dst, E0, N, H, C, esc, emax);
    k_edge_exp<<<nblk((long long)(E0 + N) * H, 256), 256, 0, stream>>>(
        dst, E0, N, H, esc, emax, denom);
    hipMemsetAsync(agg, 0, (size_t)N * HC * 4, stream);
    k_edge_agg<<<nblk((long long)(E0 + N) * HC, 256), 256, 0, stream>>>(
        gemmC, esc, denom, src, dst, E0, N, H, C, agg);
    k_finalize<<<nblk((long long)N * HC, 256), 256, 0, stream>>>(
        agg, deg, bias, hout, N, HC, relu_out);
  };

  // layer 1: BN0 -> GATv2(H=2,C=512) -> [N,1024]
  dense(x, F, KP1, bn0_g, bn0_b, 0, W1l, W1r, 1024);
  gat(a1, b1, 2, 512, hcur, 0);
  // layer 2: relu(BN1) -> GATv2(H=1,C=512) -> [N,512]
  dense(hcur, 1024, 1024, bn1_g, bn1_b, 1, W2l, W2r, 512);
  gat(a2, b2, 1, 512, hcur, 0);
  // layer 3
  dense(hcur, 512, 512, bn2_g, bn2_b, 1, W3l, W3r, 512);
  gat(a3, b3, 1, 512, hcur, 0);
  // layer 4 (relu on GAT output, no BN after)
  dense(hcur, 512, 512, bn3_g, bn3_b, 1, W4l, W4r, 512);
  gat(a4, b4, 1, 512, hcur, 1);

  // head: sigmoid(h @ Wh + bh), then gather (pred[train_idx], y[train_idx])
  k_head<<<nblk((long long)N * 32, 256), 256, 0, stream>>>(hcur, Wh, bh, pred, N, 512);
  k_gather<<<nblk(NT, 256), 256, 0, stream>>>(pred, y, tidx, (float*)d_out, NT);
}
